// SelfAttention_20581483282977
// MI455X (gfx1250) — matrix operand order repaired and checked by reading, not compiled, not run on hardware
//
#include <hip/hip_runtime.h>

// ---------------------------------------------------------------------------
// MI455X (gfx1250) implementation.
// L=512, B=32, D=512, H=512.  All GEMMs run on v_wmma_f32_16x16x32_bf16
// (bf16 inputs, fp32 accumulate).  Weights are converted to bf16 once and
// live in the 192MB L2.  The bidirectional GRU recurrence runs as a
// persistent kernel: 2 dirs x 8 workgroups, per-step device-scope spin
// barrier (atomic + s_sleep), hidden state double-buffered in global/L2.
// Loop nests keep one accumulator per N-tile live with K outermost so the
// compiler never hoists all A-fragments (no scratch spills in hot loops).
// ---------------------------------------------------------------------------

typedef __attribute__((ext_vector_type(16))) __bf16       v16bf;
typedef __attribute__((ext_vector_type(8)))  float        v8f;
typedef __attribute__((ext_vector_type(8)))  unsigned int v8u;

__device__ __forceinline__ unsigned short f2bf(float f) {
  unsigned u = __builtin_bit_cast(unsigned, f);
  u += 0x7FFFu + ((u >> 16) & 1u);           // round-to-nearest-even
  return (unsigned short)(u >> 16);
}
__device__ __forceinline__ float bf2f(unsigned short h) {
  return __builtin_bit_cast(float, (unsigned)h << 16);
}
__device__ __forceinline__ float sigmoidf(float x) {
  return 1.0f / (1.0f + __expf(-x));
}

// A fragment (16x32 bf16, M x K).  ISA layout: lanes 0-15 hold M=lane,
// K pairs {0..7,16..23}; lanes 16-31 hold M=lane-16, K pairs {8..15,24..31}.
__device__ __forceinline__ v16bf load_a_frag(const unsigned short* base,
                                             int row, int ld, int k0) {
  const int lane = threadIdx.x & 31;
  const unsigned short* p = base + row * ld + k0 + ((lane >> 4) << 3);
  v8u u;
#pragma unroll
  for (int q = 0; q < 8; ++q) {
    int koff = ((q >> 2) * 16) + ((q & 3) * 2);
    u[q] = *(const unsigned int*)(p + koff);
  }
  return __builtin_bit_cast(v16bf, u);
}

// B fragment (32x16 bf16, K x N). Column n == weight row n (contiguous K):
// lanes 0-15: N=lane, K {0..7,16..23}; lanes 16-31: N=lane-16, K {8..15,24..31}.
// Caller passes wrow = rowBase + (lane & 15).
__device__ __forceinline__ v16bf load_b_frag(const unsigned short* base,
                                             int wrow, int ld, int k0) {
  const int lane = threadIdx.x & 31;
  const unsigned short* p = base + (size_t)wrow * ld + k0 + ((lane >> 4) << 3);
  uint4 a = *(const uint4*)(p);
  uint4 b = *(const uint4*)(p + 16);
  v8u u;
  u[0] = a.x; u[1] = a.y; u[2] = a.z; u[3] = a.w;
  u[4] = b.x; u[5] = b.y; u[6] = b.z; u[7] = b.w;
  return __builtin_bit_cast(v16bf, u);
}

__device__ __forceinline__ v8f wmma_bf16(v16bf a, v16bf b, v8f c) {
  return __builtin_amdgcn_wmma_f32_16x16x32_bf16(false, a, false, b,
                                                 (short)0, c, false, false);
}

// --------------------------- prep kernels ----------------------------------

__global__ void cvt_bf16_kernel(const float* __restrict__ in,
                                unsigned short* __restrict__ out, int n) {
  int i = blockIdx.x * 256 + threadIdx.x;
  if (i < n) out[i] = f2bf(in[i]);
}

__global__ void cvt_add_bf16_kernel(const float* __restrict__ a,
                                    const float* __restrict__ b,
                                    unsigned short* __restrict__ out, int n) {
  int i = blockIdx.x * 256 + threadIdx.x;
  if (i < n) out[i] = f2bf(a[i] + b[i]);
}

// Pack [w_ih | w_hh] -> (layer,dir) x 1536 x 1536 bf16, rows = gate index.
__global__ void pack_wcat_kernel(const float* __restrict__ wih0, const float* __restrict__ whh0,
                                 const float* __restrict__ wih1, const float* __restrict__ whh1,
                                 const float* __restrict__ wih2, const float* __restrict__ whh2,
                                 unsigned short* __restrict__ wcat) {
  const int ld = blockIdx.y;            // layer*2 + dir, 0..5
  const int layer = ld >> 1, dirq = ld & 1;
  const float* wih = (layer == 0) ? wih0 : (layer == 1 ? wih1 : wih2);
  const float* whh = (layer == 0) ? whh0 : (layer == 1 ? whh1 : whh2);
  size_t i = (size_t)blockIdx.x * 256 + threadIdx.x;  // over 1536*1536
  int g = (int)(i / 1536), k = (int)(i % 1536);
  float v = (k < 1024) ? wih[((size_t)dirq * 1536 + g) * 1024 + k]
                       : whh[((size_t)dirq * 1536 + g) * 512 + (k - 1024)];
  wcat[(size_t)ld * 1536 * 1536 + i] = f2bf(v);
}

// --------------------------- attention score -------------------------------
// s[row] = sum_h tanh( sum_d p[row,d] * W12[h,d] ) * V[h],  row = l*B+b.
// Each wave keeps 4 accumulator tiles live; K outermost so each A fragment
// is loaded once and fed to 4 WMMAs (no register blowup / spills).
__global__ void attn_score_kernel(const unsigned short* __restrict__ pbf,
                                  const unsigned short* __restrict__ w12,
                                  const float* __restrict__ V,
                                  float* __restrict__ s) {
  __shared__ float red[16];
  const int tid = threadIdx.x;
  if (tid < 16) red[tid] = 0.0f;
  __syncthreads();
  const int wave = tid >> 5, lane = tid & 31;
  const int mbase = blockIdx.x * 16;
  const int rowA = mbase + (lane & 15);

  v8f acc[4] = {};
  for (int kk = 0; kk < 16; ++kk) {                    // D/32
    const v16bf a = load_a_frag(pbf, rowA, 512, kk * 32);
#pragma unroll
    for (int i = 0; i < 4; ++i) {
      const int nt = wave + i * 8;                     // 0..31 (H/16)
      acc[i] = wmma_bf16(
          a, load_b_frag(w12, nt * 16 + (lane & 15), 512, kk * 32), acc[i]);
    }
  }
  float part[8];
#pragma unroll
  for (int r = 0; r < 8; ++r) part[r] = 0.0f;
#pragma unroll
  for (int i = 0; i < 4; ++i) {
    const float vc = V[(wave + i * 8) * 16 + (lane & 15)];
#pragma unroll
    for (int r = 0; r < 8; ++r) part[r] += tanhf(acc[i][r]) * vc;
  }
  const int rbase = (lane >> 4) << 3;
#pragma unroll
  for (int r = 0; r < 8; ++r) atomicAdd(&red[rbase + r], part[r]);
  __syncthreads();
  if (tid < 16) s[mbase + tid] = red[tid];
}

// ---------------- softmax over L (axis 0) + scale = L*softmax --------------
__global__ void softmax_scale_kernel(const float* __restrict__ s,
                                     float* __restrict__ scale) {
  __shared__ float red[256];
  const int b = blockIdx.x, tid = threadIdx.x;
  float m = -1e30f;
  for (int l = tid; l < 512; l += 256) m = fmaxf(m, s[l * 32 + b]);
  red[tid] = m; __syncthreads();
  for (int off = 128; off > 0; off >>= 1) {
    if (tid < off) red[tid] = fmaxf(red[tid], red[tid + off]);
    __syncthreads();
  }
  const float mx = red[0]; __syncthreads();
  float sum = 0.0f;
  for (int l = tid; l < 512; l += 256) sum += __expf(s[l * 32 + b] - mx);
  red[tid] = sum; __syncthreads();
  for (int off = 128; off > 0; off >>= 1) {
    if (tid < off) red[tid] += red[tid + off];
    __syncthreads();
  }
  const float inv = 512.0f / red[0];
  for (int l = tid; l < 512; l += 256)
    scale[l * 32 + b] = __expf(s[l * 32 + b] - mx) * inv;
}

// --------------- gated = cat * sigmoid(cat @ Wg^T), cat=[p, scale*p] -------
// Each wave keeps 8 accumulator tiles live; K outermost, A from LDS loaded
// once per K-chunk and reused by 8 WMMAs.
__global__ void gate_kernel(const unsigned short* __restrict__ pbf,
                            const float* __restrict__ scale,
                            const unsigned short* __restrict__ wg,
                            unsigned short* __restrict__ gated) {
  __shared__ unsigned short cat[16][1024];
  const int tid = threadIdx.x;
  const int mbase = blockIdx.x * 16;
  for (int idx = tid; idx < 16 * 1024; idx += 256) {
    const int row = idx >> 10, k = idx & 1023;
    const int grow = mbase + row;
    const unsigned short pv = pbf[(size_t)grow * 512 + (k & 511)];
    cat[row][k] = (k < 512) ? pv : f2bf(scale[grow] * bf2f(pv));
  }
  __syncthreads();
  const int wave = tid >> 5, lane = tid & 31;
  const int rowA = lane & 15;

  v8f acc[8] = {};
  for (int kk = 0; kk < 32; ++kk) {                    // 2D/32
    const v16bf a = load_a_frag(&cat[0][0], rowA, 1024, kk * 32);
#pragma unroll
    for (int i = 0; i < 8; ++i) {
      const int nt = wave * 8 + i;                     // 0..63
      acc[i] = wmma_bf16(
          a, load_b_frag(wg, nt * 16 + (lane & 15), 1024, kk * 32), acc[i]);
    }
  }
#pragma unroll
  for (int i = 0; i < 8; ++i) {
    const int n = (wave * 8 + i) * 16 + (lane & 15);
#pragma unroll
    for (int r = 0; r < 8; ++r) {
      const int m = r + ((lane >> 4) << 3);
      const float cv = bf2f(cat[m][n]);
      gated[(size_t)(mbase + m) * 1024 + n] = f2bf(cv * sigmoidf(acc[i][r]));
    }
  }
}

// --------------------------- GRU layer (persistent) ------------------------
// grid = (8 column-WGs, 2 dirs), block = 256 (8 waves).  Each WG owns a
// 64-column slice of H; per step: gates = [x_t ; h_prev] @ Wcat^T via WMMA,
// x-part and h-part accumulated separately (n-gate needs xn + r*hn).
__global__ void gru_layer_kernel(const unsigned short* __restrict__ xin,   // (L,B,1024) bf16
                                 const unsigned short* __restrict__ wcat,  // (2,1536,1536) bf16
                                 const float* __restrict__ b_ih,           // (2,1536)
                                 const float* __restrict__ b_hh,           // (2,1536)
                                 unsigned short* __restrict__ hbuf,        // (2 parity,2 dir,32,512) bf16
                                 int* __restrict__ ctr,                    // [2]
                                 unsigned short* __restrict__ ybf,         // (L,B,1024) bf16 or null
                                 float* __restrict__ yf32) {               // (L,B,1024) f32 or null
  __shared__ float gx[32][192];
  __shared__ float gh[32][192];
  const int tid = threadIdx.x;
  const int wave = tid >> 5, lane = tid & 31;
  const int dir = blockIdx.y;
  const int colBase = blockIdx.x * 64;
  const unsigned short* W = wcat + (size_t)dir * 1536 * 1536;
  const float* bih = b_ih + dir * 1536;
  const float* bhh = b_hh + dir * 1536;
  const int mt = wave & 1;
  const int ntBase = (wave >> 1) * 3;
  const int laneRow = mt * 16 + (lane & 15);

  int wrows[3];
#pragma unroll
  for (int i = 0; i < 3; ++i) {
    const int nt = ntBase + i;
    wrows[i] = (nt >> 2) * 512 + colBase + (nt & 3) * 16 + (lane & 15);
  }

  for (int step = 0; step < 512; ++step) {
    const int t = dir ? (511 - step) : step;
    const unsigned short* xrow = xin + (size_t)t * 32 * 1024;
    const unsigned short* hrd =
        hbuf + ((size_t)(step & 1) * 2 + dir) * 32 * 512;
    unsigned short* hwr =
        hbuf + ((size_t)((step + 1) & 1) * 2 + dir) * 32 * 512;

    // prefetch next timestep's input rows into cache (global_prefetch_b8)
    if (step + 1 < 512) {
      const int tn = dir ? (510 - step) : (step + 1);
      const unsigned short* nx = xin + (size_t)tn * 32 * 1024;
      for (int off = tid * 64; off < 32 * 1024; off += 256 * 64)
        __builtin_prefetch(nx + off, 0, 0);
    }

    v8f accx[3] = {}, acch[3] = {};
    for (int kk = 0; kk < 32; ++kk) {                   // x part, K=1024
      const v16bf a = load_a_frag(xrow, laneRow, 1024, kk * 32);
#pragma unroll
      for (int i = 0; i < 3; ++i)
        accx[i] = wmma_bf16(a, load_b_frag(W, wrows[i], 1536, kk * 32), accx[i]);
    }
    if (step > 0) {
      for (int kk = 0; kk < 16; ++kk) {                 // h part, K=512
        const v16bf a = load_a_frag(hrd, laneRow, 512, kk * 32);
#pragma unroll
        for (int i = 0; i < 3; ++i)
          acch[i] = wmma_bf16(a, load_b_frag(W, wrows[i], 1536, 1024 + kk * 32),
                              acch[i]);
      }
    }
#pragma unroll
    for (int i = 0; i < 3; ++i) {
      const int col = (ntBase + i) * 16 + (lane & 15);
#pragma unroll
      for (int r = 0; r < 8; ++r) {
        const int m = mt * 16 + r + ((lane >> 4) << 3);
        gx[m][col] = accx[i][r];
        gh[m][col] = acch[i][r];
      }
    }
    __syncthreads();

    for (int idx = tid; idx < 32 * 64; idx += 256) {
      const int b = idx >> 6, jl = idx & 63;
      const int j = colBase + jl;
      const float r = sigmoidf(gx[b][jl] + gh[b][jl] + bih[j] + bhh[j]);
      const float z = sigmoidf(gx[b][64 + jl] + gh[b][64 + jl] +
                               bih[512 + j] + bhh[512 + j]);
      const float n = tanhf(gx[b][128 + jl] + bih[1024 + j] +
                            r * (gh[b][128 + jl] + bhh[1024 + j]));
      const float hp = (step > 0) ? bf2f(hrd[b * 512 + j]) : 0.0f;
      const float hn = (1.0f - z) * n + z * hp;
      hwr[b * 512 + j] = f2bf(hn);
      const size_t oidx = ((size_t)t * 32 + b) * 1024 + dir * 512 + j;
      if (yf32) yf32[oidx] = hn; else ybf[oidx] = f2bf(hn);
    }
    __syncthreads();

    // device-scope spin barrier across the 8 WGs of this direction
    if (tid == 0) {
      __threadfence();
      __hip_atomic_fetch_add(&ctr[dir], 1, __ATOMIC_RELEASE,
                             __HIP_MEMORY_SCOPE_AGENT);
      const int target = (step + 1) * 8;
      while (__hip_atomic_load(&ctr[dir], __ATOMIC_ACQUIRE,
                               __HIP_MEMORY_SCOPE_AGENT) < target)
        __builtin_amdgcn_s_sleep(2);
    }
    __syncthreads();
  }
}

__global__ void zero_h_kernel(unsigned int* __restrict__ h_u32, int nWords,
                              int* __restrict__ ctr) {
  const int i = blockIdx.x * 256 + threadIdx.x;
  if (i < nWords) h_u32[i] = 0u;
  if (i < 2) ctr[i] = 0;
}

// --------------------------------- host ------------------------------------

extern "C" void kernel_launch(void* const* d_in, const int* in_sizes, int n_in,
                              void* d_out, int out_size, void* d_ws,
                              size_t ws_size, hipStream_t stream) {
  (void)in_sizes; (void)n_in; (void)out_size; (void)ws_size;
  const float* prev = (const float*)d_in[0];
  const float* W1   = (const float*)d_in[1];
  const float* W2   = (const float*)d_in[2];
  const float* V    = (const float*)d_in[3];
  const float* Wg   = (const float*)d_in[4];
  const float* wih[3] = {(const float*)d_in[5], (const float*)d_in[9],  (const float*)d_in[13]};
  const float* whh[3] = {(const float*)d_in[6], (const float*)d_in[10], (const float*)d_in[14]};
  const float* bih[3] = {(const float*)d_in[7], (const float*)d_in[11], (const float*)d_in[15]};
  const float* bhh[3] = {(const float*)d_in[8], (const float*)d_in[12], (const float*)d_in[16]};

  char* ws = (char*)d_ws;
  auto alloc = [&](size_t bytes) -> char* {
    char* p = ws;
    ws += (bytes + 255) & ~(size_t)255;
    return p;
  };
  unsigned short* pbf  = (unsigned short*)alloc((size_t)512 * 32 * 512 * 2);
  unsigned short* w12  = (unsigned short*)alloc((size_t)512 * 512 * 2);
  unsigned short* wgbf = (unsigned short*)alloc((size_t)1024 * 1024 * 2);
  unsigned short* wcat = (unsigned short*)alloc((size_t)6 * 1536 * 1536 * 2);
  float* sbuf  = (float*)alloc((size_t)512 * 32 * 4);
  float* scale = (float*)alloc((size_t)512 * 32 * 4);
  unsigned short* bufA = (unsigned short*)alloc((size_t)512 * 32 * 1024 * 2);
  unsigned short* bufB = (unsigned short*)alloc((size_t)512 * 32 * 1024 * 2);
  unsigned short* hbuf = (unsigned short*)alloc((size_t)2 * 2 * 32 * 512 * 2);
  int* ctr = (int*)alloc(256);

  cvt_add_bf16_kernel<<<512 * 512 / 256, 256, 0, stream>>>(W1, W2, w12, 512 * 512);
  cvt_bf16_kernel<<<1024 * 1024 / 256, 256, 0, stream>>>(Wg, wgbf, 1024 * 1024);
  cvt_bf16_kernel<<<512 * 32 * 512 / 256, 256, 0, stream>>>(prev, pbf, 512 * 32 * 512);
  pack_wcat_kernel<<<dim3(1536 * 1536 / 256, 6), 256, 0, stream>>>(
      wih[0], whh[0], wih[1], whh[1], wih[2], whh[2], wcat);

  attn_score_kernel<<<512 * 32 / 16, 256, 0, stream>>>(pbf, w12, V, sbuf);
  softmax_scale_kernel<<<32, 256, 0, stream>>>(sbuf, scale);
  gate_kernel<<<512 * 32 / 16, 256, 0, stream>>>(pbf, scale, wgbf, bufA);

  const unsigned short* ins[3]  = {bufA, bufB, bufA};
  unsigned short* outsbf[3]     = {bufB, bufA, nullptr};
  for (int l = 0; l < 3; ++l) {
    zero_h_kernel<<<(2 * 2 * 32 * 512 * 2 / 4 + 255) / 256, 256, 0, stream>>>(
        (unsigned int*)hbuf, 2 * 2 * 32 * 512 * 2 / 4, ctr);
    gru_layer_kernel<<<dim3(8, 2), 256, 0, stream>>>(
        ins[l], wcat + (size_t)l * 2 * 1536 * 1536, bih[l], bhh[l], hbuf, ctr,
        outsbf[l], (l == 2) ? (float*)d_out : nullptr);
  }
}
